// Attention_78786880078278
// MI455X (gfx1250) — compile-verified, matrix operand order checked by reading, not run on hardware
//
#include <hip/hip_runtime.h>
#include <hip/hip_bf16.h>

// ---------------------------------------------------------------------------
// MI455X (gfx1250) attention layer, bf16 WMMA everywhere, fp32 accumulate.
//   x:[2,2048,2048] f32, Wq/Wk/Wv/Wo:[2048,2048] f32 -> out:[2,2048,2048] f32
// Pipeline:
//   (1) f32 -> bf16 convert (x, 4 weights)
//   (2) WMMA GEMM (async-LDS double-buffered W tiles)
//          Q = x*Wq^T (+fused RoPE) -> Qb [B,H,S,D] bf16
//          K = x*Wk^T (+fused RoPE) -> Kb [B,H,S,D] bf16
//          V = x*Wv^T (transposed)  -> Vt [B,H,D,S] bf16
//   (3) flash attention (online softmax), WMMA for QK^T and P*V -> ctx bf16
//   (4) WMMA GEMM  out = ctx*Wo^T -> f32
// ---------------------------------------------------------------------------

#define BATCH 2
#define SEQ 2048
#define HID 2048
#define HEADS 16
#define HDIM 128
#define MROWS (BATCH * SEQ)   // 4096
#define KCH (HID / 32)        // 64 k-chunks of 32

typedef __attribute__((ext_vector_type(16))) __bf16 v16bf;
typedef __attribute__((ext_vector_type(8)))  float  v8f;

// ln(10000)/64
#define ROPE_LN 0.14391157f
#define SOFTMAX_SCALE 0.08838834765f  // 1/sqrt(128)

// Load a 16x32-bf16 WMMA A/B fragment for one lane.
// `p` points at element (row_of_this_lane, k0 + (lane>>4)*8); the second
// 8-element chunk sits 16 elements further (ISA 16-bit A/B VGPR layout).
__device__ __forceinline__ v16bf load_frag16(const __bf16* p) {
    union { v16bf v; uint4 q[2]; } u;
    u.q[0] = *reinterpret_cast<const uint4*>(p);
    u.q[1] = *reinterpret_cast<const uint4*>(p + 16);
    return u.v;
}

__device__ __forceinline__ v8f wmma_bf16(v16bf a, v16bf b, v8f c) {
    return __builtin_amdgcn_wmma_f32_16x16x32_bf16(
        false, a, false, b, (short)0, c, false, false);
}

// LDS byte offset of a __shared__ object: flat shared addresses map to LDS
// via addr[31:0] (ISA 10.2 aperture rules), so truncation is exact.
__device__ __forceinline__ unsigned lds_off(const void* p) {
    return (unsigned)(uintptr_t)p;
}

// Per-lane async global->LDS 16-byte copy (ASYNCcnt-tracked DMA, no VGPR data).
__device__ __forceinline__ void async_ld_b128(unsigned lds_byte, const void* g) {
    asm volatile("global_load_async_to_lds_b128 %0, %1, off"
                 :: "v"(lds_byte), "v"((unsigned long long)(uintptr_t)g)
                 : "memory");
}

__device__ __forceinline__ void wait_async0() {
    asm volatile("s_wait_asynccnt 0x0" ::: "memory");
}

// ---------------------------------------------------------------------------
// Kernel 1: f32 -> bf16 (4 elements / thread)
// ---------------------------------------------------------------------------
__global__ void __launch_bounds__(256)
cvt_f32_bf16(const float* __restrict__ in, __bf16* __restrict__ out, int n) {
    int i = (blockIdx.x * 256 + threadIdx.x) * 4;
    if (i + 3 < n) {
        float4 f = *reinterpret_cast<const float4*>(in + i);
        out[i + 0] = (__bf16)f.x;
        out[i + 1] = (__bf16)f.y;
        out[i + 2] = (__bf16)f.z;
        out[i + 3] = (__bf16)f.w;
    }
}

// ---------------------------------------------------------------------------
// Kernel 2: GEMM  Y[m,n] = sum_k A[m,k] * W[n,k],  M=4096, N=K=2048.
// Block: 256 threads (8 waves). Wave w owns rows [m0+16w, m0+16w+16),
// columns [n0, n0+128)  (one full head -> in-lane RoPE in epilogue).
// W tile (128 rows x 32 k, bf16, 8KB) is DMA'd into a double-buffered LDS
// slab with GLOBAL_LOAD_ASYNC_TO_LDS_B128 so the copy of tile i+1 overlaps
// the 8 WMMAs of tile i.
// MODE 0: plain f32 row-major out.
// MODE 1: fused RoPE, bf16 out at [B,H,S,D].
// MODE 2: bf16 out transposed to [B,H,D,S]   (for V).
// ---------------------------------------------------------------------------
template <int MODE>
__global__ void __launch_bounds__(256)
gemm_bf16_wmma(const __bf16* __restrict__ A, const __bf16* __restrict__ W,
               void* __restrict__ out) {
    __shared__ __bf16 ldsB[2][128 * 32];   // 2 x 8 KB

    const int tid  = threadIdx.x;
    const int wave = tid >> 5;
    const int lane = tid & 31;
    const int half = lane >> 4;
    const int lm   = lane & 15;

    const int n0 = blockIdx.x * 128;
    const int m0 = blockIdx.y * 128 + wave * 16;

    // Async staging assignment: thread t copies 32 B of row (t>>1),
    // k-halves (t&1)*16 .. +16, as two per-lane B128 DMAs.
    const int srow = tid >> 1;
    const int skh  = (tid & 1) * 16;
    const __bf16* wrow = W + (size_t)(n0 + srow) * HID + skh;

    v8f acc[8];
#pragma unroll
    for (int t = 0; t < 8; ++t) acc[t] = v8f{0.f,0.f,0.f,0.f,0.f,0.f,0.f,0.f};

    const __bf16* arow = A + (size_t)(m0 + lm) * HID;

    // Prologue: kick off tile 0.
    {
        unsigned d = lds_off(&ldsB[0][srow * 32 + skh]);
        async_ld_b128(d,      wrow);
        async_ld_b128(d + 16, wrow + 8);
    }

    for (int i = 0; i < KCH; ++i) {
        // Our async writes for buf[i&1] done; barrier => everyone's are, and
        // everyone finished reading buf[(i+1)&1] in iteration i-1.
        wait_async0();
        __syncthreads();

        if (i + 1 < KCH) {
            const __bf16* src = wrow + (i + 1) * 32;
            unsigned d = lds_off(&ldsB[(i + 1) & 1][srow * 32 + skh]);
            async_ld_b128(d,      src);
            async_ld_b128(d + 16, src + 8);
        }

        const __bf16* buf = ldsB[i & 1];

        // A fragment straight from global (k-contiguous).
        const __bf16* ap = arow + i * 32 + half * 8;
        if (i + 1 < KCH) __builtin_prefetch(ap + 32, 0, 3);
        v16bf af = load_frag16(ap);

#pragma unroll
        for (int t = 0; t < 8; ++t) {
            const __bf16* bp = buf + (t * 16 + lm) * 32 + half * 8;
            v16bf bfr = load_frag16(bp);
            acc[t] = wmma_bf16(af, bfr, acc[t]);
        }
    }

    // ---------------- epilogues ----------------
    if constexpr (MODE == 0) {
        float* O = reinterpret_cast<float*>(out);
#pragma unroll
        for (int t = 0; t < 8; ++t) {
            const int n = n0 + t * 16 + lm;
#pragma unroll
            for (int r = 0; r < 8; ++r) {
                const int m = m0 + r + 8 * half;
                O[(size_t)m * HID + n] = acc[t][r];
            }
        }
    } else if constexpr (MODE == 1) {
        // RoPE: head column pairs (d, d+64) live in the SAME lane,
        // accumulators t and t+4. Fully in-lane rotation.
        __bf16* O = reinterpret_cast<__bf16*>(out);
        const int hh = n0 >> 7;  // head index
#pragma unroll
        for (int t = 0; t < 4; ++t) {
            const int j = t * 16 + lm;                 // 0..63
            const float invf = __expf(-ROPE_LN * (float)j);
#pragma unroll
            for (int r = 0; r < 8; ++r) {
                const int m   = m0 + r + 8 * half;
                const int bb  = m >> 11;               // / SEQ
                const int pos = m & (SEQ - 1);
                float sn, cs;
                __sincosf((float)pos * invf, &sn, &cs);
                const float x1 = acc[t][r];
                const float x2 = acc[t + 4][r];
                const size_t base =
                    ((size_t)(bb * HEADS + hh) * SEQ + pos) * HDIM;
                O[base + j]      = (__bf16)(x1 * cs - x2 * sn);
                O[base + j + 64] = (__bf16)(x2 * cs + x1 * sn);
            }
        }
    } else {  // MODE == 2 : V transposed to [B,H,D,S]
        __bf16* O = reinterpret_cast<__bf16*>(out);
        const int hh = n0 >> 7;
#pragma unroll
        for (int t = 0; t < 8; ++t) {
            const int d = t * 16 + lm;
#pragma unroll
            for (int r = 0; r < 8; ++r) {
                const int m   = m0 + r + 8 * half;
                const int bb  = m >> 11;
                const int pos = m & (SEQ - 1);
                O[((size_t)(bb * HEADS + hh) * HDIM + d) * SEQ + pos] =
                    (__bf16)acc[t][r];
            }
        }
    }
}

// ---------------------------------------------------------------------------
// Kernel 3: causal flash attention, one (b,h) x 128-query block per WG.
// Wave w owns 16 queries. Online softmax; P-tile transposed C->A layout
// through a per-wave LDS slab; O accumulated with WMMA against Vt.
// Wave loop counts diverge (causal), so no block-wide barriers inside.
// ---------------------------------------------------------------------------
__global__ void __launch_bounds__(256)
flash_attn_wmma(const __bf16* __restrict__ Qb, const __bf16* __restrict__ Kb,
                const __bf16* __restrict__ Vt, __bf16* __restrict__ ctx) {
    __shared__ __bf16 ldsP[8][16 * 32];   // 1 KB per wave

    const int bh   = blockIdx.y;
    const int b    = bh / HEADS;
    const int h    = bh % HEADS;
    const int wave = threadIdx.x >> 5;
    const int lane = threadIdx.x & 31;
    const int half = lane >> 4;
    const int lm   = lane & 15;

    const int q0 = blockIdx.x * 128 + wave * 16;

    // Preload the wave's 16x128 Q tile as 4 A-fragments (d-chunks of 32).
    v16bf qf[4];
#pragma unroll
    for (int dc = 0; dc < 4; ++dc) {
        const __bf16* qp =
            Qb + ((size_t)bh * SEQ + (q0 + lm)) * HDIM + dc * 32 + half * 8;
        qf[dc] = load_frag16(qp);
    }

    v8f o[8];
#pragma unroll
    for (int t = 0; t < 8; ++t) o[t] = v8f{0.f,0.f,0.f,0.f,0.f,0.f,0.f,0.f};
    float mrow[8], lrow[8];
#pragma unroll
    for (int r = 0; r < 8; ++r) { mrow[r] = -3.0e38f; lrow[r] = 0.f; }

    const int nChunks = (q0 + 15) / 32 + 1;   // causal bound for this wave

    for (int c = 0; c < nChunks; ++c) {
        const int kb = c * 32;

        // ---- S = Q * K^T  (16x32 tile, 8 WMMAs) ----
        v8f st[2];
        st[0] = v8f{0.f,0.f,0.f,0.f,0.f,0.f,0.f,0.f};
        st[1] = v8f{0.f,0.f,0.f,0.f,0.f,0.f,0.f,0.f};
#pragma unroll
        for (int dc = 0; dc < 4; ++dc) {
#pragma unroll
            for (int nt = 0; nt < 2; ++nt) {
                const int key = kb + nt * 16 + lm;
                const __bf16* kp =
                    Kb + ((size_t)bh * SEQ + key) * HDIM + dc * 32 + half * 8;
                if (dc == 0) __builtin_prefetch(kp + 64, 0, 3);
                v16bf kf = load_frag16(kp);
                st[nt] = wmma_bf16(qf[dc], kf, st[nt]);
            }
        }

        // ---- scale + causal mask + online softmax (rows split per half) ----
#pragma unroll
        for (int r = 0; r < 8; ++r) {
            const int q  = q0 + r + 8 * half;
            const int c0 = kb + lm;
            const int c1 = kb + 16 + lm;
            float v0 = st[0][r] * SOFTMAX_SCALE;
            float v1 = st[1][r] * SOFTMAX_SCALE;
            if (c0 > q) v0 = -3.0e38f;
            if (c1 > q) v1 = -3.0e38f;

            float rm = fmaxf(v0, v1);
#pragma unroll
            for (int off = 1; off < 16; off <<= 1)
                rm = fmaxf(rm, __shfl_xor(rm, off, 16));

            const float mnew  = fmaxf(mrow[r], rm);
            const float alpha = __expf(mrow[r] - mnew);
            const float p0 = __expf(v0 - mnew);
            const float p1 = __expf(v1 - mnew);

            float rs = p0 + p1;
#pragma unroll
            for (int off = 1; off < 16; off <<= 1)
                rs += __shfl_xor(rs, off, 16);

            lrow[r] = lrow[r] * alpha + rs;
            mrow[r] = mnew;
#pragma unroll
            for (int t = 0; t < 8; ++t) o[t][r] *= alpha;

            const int row = r + 8 * half;
            ldsP[wave][row * 32 + lm]      = (__bf16)p0;
            ldsP[wave][row * 32 + 16 + lm] = (__bf16)p1;
        }

        // P tile: C-layout -> A-layout via LDS (wave-private slab; the
        // compiler inserts s_wait_dscnt for the ds RAW).
        v16bf pf = load_frag16(&ldsP[wave][lm * 32 + half * 8]);

        // ---- O += P * V  (8 WMMAs over d-tiles of 16) ----
#pragma unroll
        for (int t = 0; t < 8; ++t) {
            const int d = t * 16 + lm;
            const __bf16* vp =
                Vt + ((size_t)bh * HDIM + d) * SEQ + kb + half * 8;
            v16bf vf = load_frag16(vp);
            o[t] = wmma_bf16(pf, vf, o[t]);
        }
    }

    // ---- normalize + write ctx [B,S,HID] bf16 ----
#pragma unroll
    for (int r = 0; r < 8; ++r) {
        const float inv_l = 1.0f / lrow[r];
        const int q = q0 + r + 8 * half;
#pragma unroll
        for (int t = 0; t < 8; ++t) {
            const int d = t * 16 + lm;
            ctx[(size_t)(b * SEQ + q) * HID + h * HDIM + d] =
                (__bf16)(o[t][r] * inv_l);
        }
    }
}

// ---------------------------------------------------------------------------
// Workspace layout (bytes), total ~112 MB
// ---------------------------------------------------------------------------
#define WS_XB   ((size_t)0)                        // 16 MB  x bf16
#define WS_WQ   ((size_t)16 * 1024 * 1024)         //  8 MB
#define WS_WK   ((size_t)24 * 1024 * 1024)
#define WS_WV   ((size_t)32 * 1024 * 1024)
#define WS_WO   ((size_t)40 * 1024 * 1024)
#define WS_QB   ((size_t)48 * 1024 * 1024)         // 16 MB [B,H,S,D]
#define WS_KB   ((size_t)64 * 1024 * 1024)
#define WS_VT   ((size_t)80 * 1024 * 1024)         // 16 MB [B,H,D,S]
#define WS_CTX  ((size_t)96 * 1024 * 1024)         // 16 MB [B,S,HID]

extern "C" void kernel_launch(void* const* d_in, const int* in_sizes, int n_in,
                              void* d_out, int out_size, void* d_ws, size_t ws_size,
                              hipStream_t stream) {
    const float* x  = (const float*)d_in[0];
    const float* Wq = (const float*)d_in[1];
    const float* Wk = (const float*)d_in[2];
    const float* Wv = (const float*)d_in[3];
    const float* Wo = (const float*)d_in[4];

    char* ws = (char*)d_ws;
    __bf16* xb  = (__bf16*)(ws + WS_XB);
    __bf16* wqb = (__bf16*)(ws + WS_WQ);
    __bf16* wkb = (__bf16*)(ws + WS_WK);
    __bf16* wvb = (__bf16*)(ws + WS_WV);
    __bf16* wob = (__bf16*)(ws + WS_WO);
    __bf16* Qb  = (__bf16*)(ws + WS_QB);
    __bf16* Kb  = (__bf16*)(ws + WS_KB);
    __bf16* Vt  = (__bf16*)(ws + WS_VT);
    __bf16* ctx = (__bf16*)(ws + WS_CTX);

    const int nx = BATCH * SEQ * HID;   // 8388608
    const int nw = HID * HID;           // 4194304

    cvt_f32_bf16<<<nx / 1024, 256, 0, stream>>>(x,  xb,  nx);
    cvt_f32_bf16<<<nw / 1024, 256, 0, stream>>>(Wq, wqb, nw);
    cvt_f32_bf16<<<nw / 1024, 256, 0, stream>>>(Wk, wkb, nw);
    cvt_f32_bf16<<<nw / 1024, 256, 0, stream>>>(Wv, wvb, nw);
    cvt_f32_bf16<<<nw / 1024, 256, 0, stream>>>(Wo, wob, nw);

    dim3 ggrid(HID / 128, MROWS / 128);   // (16, 32)
    gemm_bf16_wmma<1><<<ggrid, 256, 0, stream>>>(xb, wqb, (void*)Qb);
    gemm_bf16_wmma<1><<<ggrid, 256, 0, stream>>>(xb, wkb, (void*)Kb);
    gemm_bf16_wmma<2><<<ggrid, 256, 0, stream>>>(xb, wvb, (void*)Vt);

    dim3 agrid(SEQ / 128, BATCH * HEADS); // (16, 32)
    flash_attn_wmma<<<agrid, 256, 0, stream>>>(Qb, Kb, Vt, ctx);

    gemm_bf16_wmma<0><<<ggrid, 256, 0, stream>>>(ctx, wob, d_out);
}
